// MultiDomainNet_72730976190754
// MI455X (gfx1250) — matrix operand order repaired and checked by reading, not compiled, not run on hardware
//
#include <hip/hip_runtime.h>
#include <hip/hip_bf16.h>

// MI455X / gfx1250, wave32. D = A*B + C via v_wmma_f32_16x16x32_f16,
// plus TDM (tensor_load_to_lds) for the h-matrix stage in kernel B.

typedef __attribute__((ext_vector_type(16))) _Float16 v16h;
typedef __attribute__((ext_vector_type(8)))  _Float16 v8h;
typedef __attribute__((ext_vector_type(4)))  _Float16 v4h;
typedef __attribute__((ext_vector_type(8)))  float    v8f;
typedef __attribute__((ext_vector_type(4)))  unsigned v4u;
typedef __attribute__((ext_vector_type(8)))  unsigned v8u;

#define NB   2
#define NAA  20
#define NF   12
#define ND   64
#define NS   512
#define NC   1024
#define NR   2048   // 2*NC

// Combine two contiguous 8-half LDS vectors (16B aligned each) into a v16h frag.
__device__ __forceinline__ v16h ld8x2(const _Float16* p0, const _Float16* p1) {
  v8h lo = *(const v8h*)p0;
  v8h hi = *(const v8h*)p1;
  v16h r;
#pragma unroll
  for (int i = 0; i < 8; ++i) { r[i] = lo[i]; r[i + 8] = hi[i]; }
  return r;
}

__global__ void mdn_zero_deg(float* deg, int n) {
  int i = blockIdx.x * 256 + threadIdx.x;
  if (i < n) deg[i] = 0.0f;
}

// ---------------------------------------------------------------------------
// Kernel A: per (domain, 128-row r-chunk):
//   x[m, r] = sum_s aa16[m, s] * adj16[r, s]   (M=48 pad of 40, N=128, K=512)
//   then conv (k=2, stride 2):  h[b,f,p] = sum_{c,k} x[b*20+c, 2p+k] * wconv[f,c,k]
//   plus partial deg[s] += sum over this chunk's 128 rows.
// LDS carve (dynamic, 64128 B total):
//   [0)      aa16  48 x 520 f16  (49920 B)   <- overlapped later by xS 48x132 f32
//   [49920)  btS  128 x  40 f16  (10240 B)
//   [60160)  degS 512 f32        ( 2048 B)
//   [62208)  wcS  480 f32        ( 1920 B)
// ---------------------------------------------------------------------------
__global__ void mdn_expand_conv(const float* __restrict__ aa,
                                const float* __restrict__ adjs,
                                const float* __restrict__ wconv,
                                _Float16* __restrict__ hbuf,
                                float* __restrict__ deg) {
  extern __shared__ char smem[];
  _Float16* aaS  = (_Float16*)(smem);
  _Float16* btS  = (_Float16*)(smem + 49920);
  float*    xS   = (float*)(smem);            // overlaps aaS (used after K loop)
  float*    degS = (float*)(smem + 60160);
  float*    wcS  = (float*)(smem + 62208);

  const int tid    = threadIdx.x;
  const int d      = blockIdx.y;
  const int rchunk = blockIdx.x;        // 0..15
  const int r0     = rchunk * 128;
  const int wv     = tid >> 5;          // wave 0..7 -> N strip
  const int lane   = tid & 31;
  const int lhalf  = lane >> 4;
  const int l16    = lane & 15;

  for (int i = tid; i < 512; i += 256) degS[i] = 0.0f;
  for (int i = tid; i < NF * NAA * 2; i += 256) wcS[i] = wconv[i];

  // Stage aa -> f16, rows m=b*20+c (48 rows, rows >=40 zero), row stride 520.
  for (int i4 = tid; i4 < 48 * 128; i4 += 256) {
    int m = i4 >> 7, s4 = i4 & 127;
    float4 v = make_float4(0.f, 0.f, 0.f, 0.f);
    if (m < NB * NAA)
      v = *(const float4*)(aa + (size_t)m * (ND * NS) + (size_t)d * NS + s4 * 4);
    _Float16* dst = aaS + m * 520 + s4 * 4;
    dst[0] = (_Float16)v.x; dst[1] = (_Float16)v.y;
    dst[2] = (_Float16)v.z; dst[3] = (_Float16)v.w;
  }
  __syncthreads();

  v8f acc[3];
#pragma unroll
  for (int t = 0; t < 3; ++t) acc[t] = (v8f){0.f,0.f,0.f,0.f,0.f,0.f,0.f,0.f};

  const size_t adjBase = (size_t)d * ((size_t)NR * NS);
  const int q    = tid & 7;   // s-quad within 32-wide K window
  const int jrow = tid >> 3;  // base row 0..31

  for (int kk = 0; kk < 16; ++kk) {          // K = 512 in steps of 32
    __syncthreads();                          // btS safe to overwrite
    // Stage 128 rows x 32 s of adjs -> f16 B tile; accumulate deg partials.
    float dacc[4] = {0.f, 0.f, 0.f, 0.f};
#pragma unroll
    for (int it = 0; it < 4; ++it) {
      int j = jrow + it * 32;
      const float* gp = adjs + adjBase + (size_t)(r0 + j) * NS + kk * 32 + q * 4;
      float4 v = *(const float4*)gp;
      _Float16* dst = btS + j * 40 + q * 4;
      dst[0] = (_Float16)v.x; dst[1] = (_Float16)v.y;
      dst[2] = (_Float16)v.z; dst[3] = (_Float16)v.w;
      dacc[0] += v.x; dacc[1] += v.y; dacc[2] += v.z; dacc[3] += v.w;
    }
#pragma unroll
    for (int i = 0; i < 4; ++i)
      atomicAdd(&degS[kk * 32 + q * 4 + i], dacc[i]);     // ds_add_f32
    if (kk < 15)                                          // global_prefetch_b8
      __builtin_prefetch(adjs + adjBase + (size_t)(r0 + jrow) * NS + (kk + 1) * 32 + q * 4, 0, 1);
    __syncthreads();

    // B fragment: column n = r-local, lane<16 holds K {0..7,16..23}, lane>=16 K {8..15,24..31}.
    const _Float16* bp = btS + (wv * 16 + l16) * 40 + lhalf * 8;
    v16h bf = ld8x2(bp, bp + 16);

    // A fragments: lane<16 K {0..7,16..23}, lane>=16 K {8..15,24..31}.
#pragma unroll
    for (int t = 0; t < 3; ++t) {
      const _Float16* ap = aaS + (t * 16 + l16) * 520 + kk * 32 + lhalf * 8;
      v16h af = ld8x2(ap, ap + 16);
      acc[t] = __builtin_amdgcn_wmma_f32_16x16x32_f16(
                 false, af, false, bf, (short)0, acc[t], false, false);
    }
  }

  __syncthreads();   // all frag reads of aaS done before overlaying with xS
  // Spill x tile: element acc[t][i] at lane L -> row t*16+i+8*(L>>4), col wv*16+(L&15)
#pragma unroll
  for (int t = 0; t < 3; ++t)
#pragma unroll
    for (int i = 0; i < 8; ++i)
      xS[(t * 16 + i + 8 * lhalf) * 132 + wv * 16 + l16] = acc[t][i];
  __syncthreads();

  // Tiny conv: h[b,f,p] for 64 p values of this chunk (r = 2p, 2p+1).
  for (int idx = tid; idx < NB * NF * 64; idx += 256) {
    int p = idx & 63;
    int f = (idx >> 6) % NF;
    int b = idx / (64 * NF);
    float s = 0.0f;
#pragma unroll
    for (int c = 0; c < NAA; ++c) {
      s += xS[(b * NAA + c) * 132 + 2 * p]     * wcS[(f * NAA + c) * 2]
         + xS[(b * NAA + c) * 132 + 2 * p + 1] * wcS[(f * NAA + c) * 2 + 1];
    }
    hbuf[(size_t)d * (24 * NC) + (size_t)(b * NF + f) * NC + rchunk * 64 + p] = (_Float16)s;
  }
  // Publish this chunk's deg partials (16 chunks/domain accumulate).
  for (int s = tid; s < NS; s += 256)
    atomicAdd(&deg[d * NS + s], degS[s]);                 // global_atomic_add_f32
}

// ---------------------------------------------------------------------------
// Kernel B: per domain:
//   y[m, s] = sum_p h16[m, p] * adjE16[p, s]  (M=24 pad 32, K=1024, N=512)
//   pooled[m] = max_s y[m,s]/deg[s];  out[d*2+b] = sigmoid(sum_f pooled*wcomb)
// h is staged into LDS by the Tensor Data Mover (tensor_load_to_lds) with
// TDM padding: 4 DWORDs (16 B = 8 halves) inserted every 256 DWORDs (512
// halves), giving a row pitch of 1040 halves with a mid-row pad at k=512.
// LDS carve (61312 B):
//   [0)      hS   24 rows x 2080 B (49920 B, TDM-padded)
//   [49920)  btS  128 x  40 f16    (10240 B)   transposed [s_local][k]
//   [60160)  redS 8 x 32 f32       ( 1024 B)
//   [61184)  poolS 32 f32          (  128 B)
// ---------------------------------------------------------------------------
__global__ void mdn_contract_pool(const float* __restrict__ adjs,
                                  const _Float16* __restrict__ hbuf,
                                  const float* __restrict__ deg,
                                  const float* __restrict__ wcomb,
                                  float* __restrict__ out) {
  extern __shared__ char smem[];
  _Float16* hS    = (_Float16*)(smem);
  _Float16* btS   = (_Float16*)(smem + 49920);
  float*    redS  = (float*)(smem + 60160);
  float*    poolS = (float*)(smem + 61184);

  const int tid   = threadIdx.x;
  const int d     = blockIdx.x;
  const int wv    = tid >> 5;
  const int lane  = tid & 31;
  const int lhalf = lane >> 4;
  const int l16   = lane & 15;

  // --- TDM stage of h: 2D tile 1024(x) x 24(y) of 2-byte elements, wave 0 only.
  if (wv == 0) {
    unsigned long long ga = (unsigned long long)(const void*)(hbuf + (size_t)d * (24 * NC));
    // D# group 0: count=1 | lds_addr=0 | global_addr | type=2
    v4u g0;
    g0[0] = 1u;                                   // count=1, user mode
    g0[1] = 0u;                                   // lds_addr (bytes): carve base
    g0[2] = (unsigned)(ga & 0xFFFFFFFFu);         // global_addr[31:0]
    g0[3] = (unsigned)((ga >> 32) & 0x01FFFFFFu) | (2u << 30);  // addr[56:32] | type=2
    // D# group 1: data_size=2B, pad_enable, pad_interval=7 (256 DW), pad_amount=3 (4 DW)
    v8u g1;
    g1[0] = (1u << 16) | (1u << 20) | (7u << 22) | (3u << 25);
    g1[1] = (unsigned)NC << 16;                   // tensor_dim0[15:0] = 1024
    g1[2] = 24u << 16;                            // tensor_dim1[15:0] = 24
    g1[3] = (unsigned)NC << 16;                   // tile_dim0 = 1024
    g1[4] = 24u;                                  // tile_dim1 = 24, tile_dim2 = 0
    g1[5] = (unsigned)NC;                         // tensor_dim0_stride[31:0] = 1024
    g1[6] = 0u;                                   // stride hi, dim1_stride lo
    g1[7] = 0u;                                   // dim1_stride hi
    asm volatile("tensor_load_to_lds %0, %1" :: "s"(g0), "s"(g1) : "memory");
    __builtin_amdgcn_s_wait_tensorcnt(0);         // s_wait_tensorcnt 0x0
  }
  redS[tid] = -3.4e38f;
  const size_t adjBase = (size_t)d * ((size_t)NR * NS);

  for (int sc = 0; sc < 4; ++sc) {           // four 128-wide s chunks
    v8f acc[2];
#pragma unroll
    for (int t = 0; t < 2; ++t) acc[t] = (v8f){0.f,0.f,0.f,0.f,0.f,0.f,0.f,0.f};

    for (int kk = 0; kk < 32; ++kk) {        // K = 1024 in steps of 32
      __syncthreads();                       // (first pass also publishes TDM data)
      // Stage adjs even rows 32(p) x 128(s), stored transposed [s_local][k].
#pragma unroll
      for (int it = 0; it < 4; ++it) {
        int idx = tid + it * 256;
        int pk = idx >> 5;                   // 0..31 within K window
        int qc = idx & 31;                   // s quad
        const float* gp = adjs + adjBase
                        + (size_t)(2 * (kk * 32 + pk)) * NS + sc * 128 + qc * 4;
        float4 v = *(const float4*)gp;
        btS[(qc * 4 + 0) * 40 + pk] = (_Float16)v.x;
        btS[(qc * 4 + 1) * 40 + pk] = (_Float16)v.y;
        btS[(qc * 4 + 2) * 40 + pk] = (_Float16)v.z;
        btS[(qc * 4 + 3) * 40 + pk] = (_Float16)v.w;
      }
      __syncthreads();

      const _Float16* bp = btS + (wv * 16 + l16) * 40 + lhalf * 8;
      v16h bf = ld8x2(bp, bp + 16);

      // TDM-padded h row pitch: 1040 halves; +8 halves once k >= 512.
      const int kbase = kk * 32 + lhalf * 8 + ((kk >= 16) ? 8 : 0);
#pragma unroll
      for (int t = 0; t < 2; ++t) {
        int m = t * 16 + l16;
        v16h af;
#pragma unroll
        for (int i = 0; i < 16; ++i) af[i] = (_Float16)0.f;
        if (m < 24) {                        // rows 24..31 are zero padding
          const _Float16* ap = hS + m * 1040 + kbase;
          af = ld8x2(ap, ap + 16);
        }
        acc[t] = __builtin_amdgcn_wmma_f32_16x16x32_f16(
                   false, af, false, bf, (short)0, acc[t], false, false);
      }
    }

    // Degree-normalized max over this chunk's 16 s columns per wave strip.
    int s = sc * 128 + wv * 16 + l16;
    float inv = 1.0f / deg[d * NS + s];
#pragma unroll
    for (int t = 0; t < 2; ++t)
#pragma unroll
      for (int i = 0; i < 8; ++i) {
        float v = acc[t][i] * inv;
#pragma unroll
        for (int mk = 1; mk < 16; mk <<= 1)
          v = fmaxf(v, __shfl_xor(v, mk, 32));  // reduce within 16-lane halves
        if (l16 == 0) {
          int m = t * 16 + i + 8 * lhalf;
          redS[wv * 32 + m] = fmaxf(redS[wv * 32 + m], v);
        }
      }
  }

  __syncthreads();
  if (tid < 32) {
    float v = redS[tid];
#pragma unroll
    for (int w = 1; w < 8; ++w) v = fmaxf(v, redS[w * 32 + tid]);
    poolS[tid] = v;
  }
  __syncthreads();
  if (tid < NB) {
    float scv = 0.0f;
#pragma unroll
    for (int f = 0; f < NF; ++f) scv += poolS[tid * NF + f] * wcomb[f];
    out[d * NB + tid] = 1.0f / (1.0f + expf(-scv));
  }
}

extern "C" void kernel_launch(void* const* d_in, const int* in_sizes, int n_in,
                              void* d_out, int out_size, void* d_ws, size_t ws_size,
                              hipStream_t stream) {
  (void)in_sizes; (void)n_in; (void)out_size; (void)ws_size;
  const float* aa    = (const float*)d_in[0];
  const float* adjs  = (const float*)d_in[1];
  const float* wconv = (const float*)d_in[2];
  const float* wcomb = (const float*)d_in[3];
  float* out = (float*)d_out;

  float*    deg  = (float*)d_ws;                               // 64*512 f32
  _Float16* hbuf = (_Float16*)((char*)d_ws + (size_t)ND * NS * sizeof(float));

  mdn_zero_deg<<<(ND * NS + 255) / 256, 256, 0, stream>>>(deg, ND * NS);

  dim3 gA(16, ND);
  mdn_expand_conv<<<gA, 256, 64128, stream>>>(aa, adjs, wconv, hbuf, deg);

  mdn_contract_pool<<<ND, 256, 61312, stream>>>(adjs, hbuf, deg, wcomb, out);
}
